// NeuronWanTransformerV3CP_2723009265850
// MI455X (gfx1250) — matrix operand order repaired and checked by reading, not compiled, not run on hardware
//
#include <hip/hip_runtime.h>
#include <stdint.h>

typedef unsigned short u16;
typedef __attribute__((ext_vector_type(16))) __bf16 v16bf;
typedef __attribute__((ext_vector_type(8)))  float  v8f;

#define S_LEN 2048
#define DIMN  2048
#define HD    128
#define NH    16

static_assert(sizeof(v16bf) == 32, "v16bf must be 32 bytes");

__device__ __forceinline__ u16 f2bf(float f) {
  union { float f; unsigned u; } c; c.f = f;
  unsigned u = c.u;
  u += 0x7FFFu + ((u >> 16) & 1u);   // round-to-nearest-even
  return (u16)(u >> 16);
}

// CDNA5 async global->LDS copy (ASYNCcnt-tracked, no VGPR round trip).
__device__ __forceinline__ void async_ld_b128(unsigned lds_off, const void* gaddr) {
  asm volatile("global_load_async_to_lds_b128 %0, %1, off"
               :: "v"(lds_off), "v"(gaddr) : "memory");
}
__device__ __forceinline__ void async_ld_b128_o16(unsigned lds_off, const void* gaddr) {
  asm volatile("global_load_async_to_lds_b128 %0, %1, off offset:16"
               :: "v"(lds_off), "v"(gaddr) : "memory");
}
__device__ __forceinline__ void wait_async0() {
  asm volatile("s_wait_asynccnt 0x0" ::: "memory");
}

// ---------------------------------------------------------------- cast f32->bf16
__global__ __launch_bounds__(256) void k_cast_bf16(const float* __restrict__ in,
                                                   u16* __restrict__ out, int n) {
  int i = (blockIdx.x * 256 + threadIdx.x) * 8;
  if (i + 7 >= n) return;
  float4 a = *(const float4*)(in + i);
  float4 b = *(const float4*)(in + i + 4);
  union { u16 s[8]; uint4 q; } r;
  r.s[0] = f2bf(a.x); r.s[1] = f2bf(a.y); r.s[2] = f2bf(a.z); r.s[3] = f2bf(a.w);
  r.s[4] = f2bf(b.x); r.s[5] = f2bf(b.y); r.s[6] = f2bf(b.z); r.s[7] = f2bf(b.w);
  *(uint4*)(out + i) = r.q;
}

// ---------------------------------------------------------------- bf16 NT GEMM + bias
// C[M,N] (f32) = A[M,K](bf16 row-major) * B[N,K](bf16 row-major)^T + bias[N]
// block tile 128x128, 256 threads = 8 waves (2 M x 4 N), wave tile 64x32.
// Staging: double-buffered LDS fed by global_load_async_to_lds_b128.
__global__ __launch_bounds__(256) void k_gemm_bf16_nt(
    const u16* __restrict__ A, const u16* __restrict__ B,
    const float* __restrict__ bias, float* __restrict__ C,
    int M, int N, int K) {
  constexpr int LRS = 40;                       // LDS row stride in u16 (80 B, conflict-free)
  __shared__ u16 lA[2][128 * LRS];
  __shared__ u16 lB[2][128 * LRS];
  const int t = threadIdx.x;
  const int lane = t & 31, w = t >> 5;
  const int wm = (w >> 2) * 64, wn = (w & 3) * 32;
  const int bm = blockIdx.y * 128, bn = blockIdx.x * 128;
  const int lrow = t >> 1, lcol = (t & 1) * 16; // staging: 16 bf16 per thread
  const int lh = lane >> 4, ll = lane & 15;

  v8f zero = {0.f,0.f,0.f,0.f,0.f,0.f,0.f,0.f};
  v8f acc[4][2];
#pragma unroll
  for (int mi = 0; mi < 4; ++mi)
#pragma unroll
    for (int ni = 0; ni < 2; ++ni) acc[mi][ni] = zero;

  const u16* gA = A + (size_t)(bm + lrow) * K + lcol;
  const u16* gB = B + (size_t)(bn + lrow) * K + lcol;

  // per-thread LDS staging destinations (byte offsets in the LDS aperture)
  unsigned stA[2], stB[2];
  stA[0] = (unsigned)(size_t)&lA[0][lrow * LRS + lcol];
  stA[1] = (unsigned)(size_t)&lA[1][lrow * LRS + lcol];
  stB[0] = (unsigned)(size_t)&lB[0][lrow * LRS + lcol];
  stB[1] = (unsigned)(size_t)&lB[1][lrow * LRS + lcol];

  union Frag { uint4 q[2]; v16bf v; };

  // prologue: stage tile 0 into buffer 0
  async_ld_b128(stA[0], gA);
  async_ld_b128_o16(stA[0] + 16, gA);
  async_ld_b128(stB[0], gB);
  async_ld_b128_o16(stB[0] + 16, gB);
  gA += 32; gB += 32;
  wait_async0();
  __syncthreads();

  int buf = 0;
  for (int k0 = 0; k0 < K; k0 += 32) {
    const int nxt = buf ^ 1;
    if (k0 + 32 < K) {                          // stage next tile while computing
      async_ld_b128(stA[nxt], gA);
      async_ld_b128_o16(stA[nxt] + 16, gA);
      async_ld_b128(stB[nxt], gB);
      async_ld_b128_o16(stB[nxt] + 16, gB);
      gA += 32; gB += 32;
    }

    const u16* bufA = lA[buf];
    const u16* bufB = lB[buf];
    Frag fa[4], fb[2];
#pragma unroll
    for (int mi = 0; mi < 4; ++mi) {
      int r = wm + mi * 16 + ll;
      fa[mi].q[0] = *(const uint4*)&bufA[r * LRS + lh * 8];        // K 0..7 | 8..15
      fa[mi].q[1] = *(const uint4*)&bufA[r * LRS + 16 + lh * 8];   // K 16..23 | 24..31
    }
#pragma unroll
    for (int ni = 0; ni < 2; ++ni) {
      int c = wn + ni * 16 + ll;
      fb[ni].q[0] = *(const uint4*)&bufB[c * LRS + lh * 8];        // K 0..7 | 8..15
      fb[ni].q[1] = *(const uint4*)&bufB[c * LRS + 16 + lh * 8];   // K 16..23 | 24..31
    }
#pragma unroll
    for (int mi = 0; mi < 4; ++mi)
#pragma unroll
      for (int ni = 0; ni < 2; ++ni)
        acc[mi][ni] = __builtin_amdgcn_wmma_f32_16x16x32_bf16(
            false, fa[mi].v, false, fb[ni].v, (short)0, acc[mi][ni], false, false);

    wait_async0();                               // next tile fully in LDS (this wave)
    __syncthreads();                             // ... and for all waves
    buf = nxt;
  }

#pragma unroll
  for (int mi = 0; mi < 4; ++mi)
#pragma unroll
    for (int ni = 0; ni < 2; ++ni) {
      int col = bn + wn + ni * 16 + ll;
      float bv = bias[col];
      int rb = bm + wm + mi * 16 + lh * 8;
#pragma unroll
      for (int r = 0; r < 8; ++r)
        C[(size_t)(rb + r) * N + col] = acc[mi][ni][r] + bv;
    }
}

// ---------------------------------------------------------------- RMSNorm + RoPE + cast
// in: X[S,DIM] f32; out: [NH][S][HD] bf16
__global__ __launch_bounds__(256) void k_rmsnorm_rope(
    const float* __restrict__ X, const float* __restrict__ nw,
    const float* __restrict__ fc, const float* __restrict__ fs,
    u16* __restrict__ out) {
  int s = blockIdx.x;
  int t = threadIdx.x;
  int idx = t * 8;
  const float* row = X + (size_t)s * DIMN;
  float v[8];
  { float4 x0 = *(const float4*)(row + idx);
    float4 x1 = *(const float4*)(row + idx + 4);
    v[0]=x0.x; v[1]=x0.y; v[2]=x0.z; v[3]=x0.w;
    v[4]=x1.x; v[5]=x1.y; v[6]=x1.z; v[7]=x1.w; }
  float ss = 0.f;
#pragma unroll
  for (int i = 0; i < 8; ++i) ss += v[i] * v[i];
  for (int off = 16; off > 0; off >>= 1) ss += __shfl_xor(ss, off, 32);
  __shared__ float red[8];
  if ((t & 31) == 0) red[t >> 5] = ss;
  __syncthreads();
  float tot = 0.f;
#pragma unroll
  for (int i = 0; i < 8; ++i) tot += red[i];
  float inv = rsqrtf(tot * (1.0f / DIMN) + 1e-6f);

  float wv[8];
  { float4 w0 = *(const float4*)(nw + idx);
    float4 w1 = *(const float4*)(nw + idx + 4);
    wv[0]=w0.x; wv[1]=w0.y; wv[2]=w0.z; wv[3]=w0.w;
    wv[4]=w1.x; wv[5]=w1.y; wv[6]=w1.z; wv[7]=w1.w; }
  float y[8];
#pragma unroll
  for (int i = 0; i < 8; ++i) y[i] = v[i] * inv * wv[i];

  int dl = idx & 127;                    // dim-in-head, even, 8 consecutive stay in one head
  const float* fcs = fc + (size_t)s * HD;
  const float* fss = fs + (size_t)s * HD;
  union { u16 u[8]; uint4 q; } r;
#pragma unroll
  for (int j = 0; j < 4; ++j) {
    float c  = fcs[dl + 2 * j];          // freqs_cos[..., 0::2][pair]
    float sn = fss[dl + 2 * j + 1];      // freqs_sin[..., 1::2][pair]
    float e = y[2 * j], o = y[2 * j + 1];
    r.u[2 * j]     = f2bf(e * c - o * sn);
    r.u[2 * j + 1] = f2bf(e * sn + o * c);
  }
  int h = idx >> 7;
  *(uint4*)(out + (size_t)h * S_LEN * HD + (size_t)s * HD + dl) = r.q;
}

// ---------------------------------------------------------------- V transpose-cast
// in: V[S,DIM] f32 ; out: Vt[NH][HD][S] bf16
__global__ __launch_bounds__(256) void k_vtrans(const float* __restrict__ V,
                                                u16* __restrict__ Vt) {
  int d0 = blockIdx.x * 64;
  int s0 = blockIdx.y * 64;
  int h  = blockIdx.z;
  __shared__ float tile[64 * 68];
  int t = threadIdx.x;
  int sr = t >> 2, cc = (t & 3) * 16;
  const float* src = V + (size_t)(s0 + sr) * DIMN + h * HD + d0 + cc;
#pragma unroll
  for (int i = 0; i < 4; ++i)
    *(float4*)&tile[sr * 68 + cc + 4 * i] = *(const float4*)(src + 4 * i);
  __syncthreads();
  int dr = t >> 2, sc = (t & 3) * 16;
  union { u16 u[8]; uint4 q; } r0, r1;
#pragma unroll
  for (int i = 0; i < 8; ++i) r0.u[i] = f2bf(tile[(sc + i) * 68 + dr]);
#pragma unroll
  for (int i = 0; i < 8; ++i) r1.u[i] = f2bf(tile[(sc + 8 + i) * 68 + dr]);
  u16* dst = Vt + (size_t)h * HD * S_LEN + (size_t)(d0 + dr) * S_LEN + s0 + sc;
  *(uint4*)(dst)     = r0.q;
  *(uint4*)(dst + 8) = r1.q;
}

// ---------------------------------------------------------------- flash attention
// Q,K: [NH][S][HD] bf16 ; Vt: [NH][HD][S] bf16 ; O: [S][DIM] bf16
// block = 4 waves, 64 query rows; wave = 16 query rows; kv tile = 64.
__global__ __launch_bounds__(128) void k_flash(
    const u16* __restrict__ Q, const u16* __restrict__ Kb,
    const u16* __restrict__ Vt, u16* __restrict__ O) {
  const int h = blockIdx.y;
  const int q0 = blockIdx.x * 64;
  const int t = threadIdx.x, lane = t & 31, w = t >> 5;
  const int lh = lane >> 4, ll = lane & 15;
  const int qb = q0 + w * 16;
  __shared__ u16 lp[4 * 16 * 72];       // per-wave P tile, row stride 72 u16 (144 B)
  u16* pw = lp + w * 16 * 72;

  const u16* Qh = Q  + (size_t)h * S_LEN * HD;
  const u16* Kh = Kb + (size_t)h * S_LEN * HD;
  const u16* Vh = Vt + (size_t)h * HD * S_LEN;

  union Frag { uint4 q[2]; v16bf v; };
  Frag qa[4];
#pragma unroll
  for (int kd = 0; kd < 4; ++kd) {
    const u16* p = Qh + (size_t)(qb + ll) * HD + kd * 32;
    qa[kd].q[0] = *(const uint4*)(p + lh * 8);
    qa[kd].q[1] = *(const uint4*)(p + 16 + lh * 8);
  }

  v8f zero = {0.f,0.f,0.f,0.f,0.f,0.f,0.f,0.f};
  v8f o[8];
#pragma unroll
  for (int dt = 0; dt < 8; ++dt) o[dt] = zero;
  float mr[8], lr[8];
#pragma unroll
  for (int e = 0; e < 8; ++e) { mr[e] = -1e30f; lr[e] = 0.f; }
  const float SL = 0.08838834764831845f * 1.4426950408889634f; // (1/sqrt(128))*log2(e)

  for (int kv = 0; kv < S_LEN; kv += 64) {
    v8f sc[4];
#pragma unroll
    for (int nt = 0; nt < 4; ++nt) {
      sc[nt] = zero;
      const u16* kp = Kh + (size_t)(kv + nt * 16 + ll) * HD;
#pragma unroll
      for (int kd = 0; kd < 4; ++kd) {
        Frag kf;
        kf.q[0] = *(const uint4*)(kp + kd * 32 + lh * 8);
        kf.q[1] = *(const uint4*)(kp + kd * 32 + 16 + lh * 8);
        sc[nt] = __builtin_amdgcn_wmma_f32_16x16x32_bf16(
            false, qa[kd].v, false, kf.v, (short)0, sc[nt], false, false);
      }
    }
    // online softmax stats (half-wave = the 16 lanes holding one row's columns)
    float alpha[8], ps[8];
#pragma unroll
    for (int e = 0; e < 8; ++e) {
      float m = fmaxf(fmaxf(sc[0][e], sc[1][e]), fmaxf(sc[2][e], sc[3][e]));
      for (int off = 1; off < 16; off <<= 1) m = fmaxf(m, __shfl_xor(m, off, 32));
      float mn = fmaxf(mr[e], m);
      alpha[e] = exp2f((mr[e] - mn) * SL);
      mr[e] = mn;
      ps[e] = 0.f;
    }
#pragma unroll
    for (int nt = 0; nt < 4; ++nt)
#pragma unroll
      for (int e = 0; e < 8; ++e) {
        float p = exp2f((sc[nt][e] - mr[e]) * SL);
        ps[e] += p;
        pw[(e + lh * 8) * 72 + nt * 16 + ll] = f2bf(p);  // C-layout -> [m][kv] LDS
      }
#pragma unroll
    for (int e = 0; e < 8; ++e) {
      float sum = ps[e];
      for (int off = 1; off < 16; off <<= 1) sum += __shfl_xor(sum, off, 32);
      lr[e] = lr[e] * alpha[e] + sum;
    }
#pragma unroll
    for (int dt = 0; dt < 8; ++dt)
#pragma unroll
      for (int e = 0; e < 8; ++e) o[dt][e] = o[dt][e] * alpha[e];

    // P @ V
    Frag pa[2];
#pragma unroll
    for (int kp2 = 0; kp2 < 2; ++kp2) {
      const u16* pp = pw + ll * 72 + kp2 * 32;
      pa[kp2].q[0] = *(const uint4*)(pp + lh * 8);
      pa[kp2].q[1] = *(const uint4*)(pp + 16 + lh * 8);
    }
#pragma unroll
    for (int dt = 0; dt < 8; ++dt) {
      const u16* vp = Vh + (size_t)(dt * 16 + ll) * S_LEN + kv;
#pragma unroll
      for (int kp2 = 0; kp2 < 2; ++kp2) {
        Frag vf;
        vf.q[0] = *(const uint4*)(vp + kp2 * 32 + lh * 8);
        vf.q[1] = *(const uint4*)(vp + kp2 * 32 + 16 + lh * 8);
        o[dt] = __builtin_amdgcn_wmma_f32_16x16x32_bf16(
            false, pa[kp2].v, false, vf.v, (short)0, o[dt], false, false);
      }
    }
  }

  float il[8];
#pragma unroll
  for (int e = 0; e < 8; ++e) il[e] = 1.f / lr[e];
#pragma unroll
  for (int dt = 0; dt < 8; ++dt)
#pragma unroll
    for (int e = 0; e < 8; ++e) {
      float val = o[dt][e] * il[e];
      O[(size_t)(qb + e + lh * 8) * DIMN + h * HD + dt * 16 + ll] = f2bf(val);
    }
}

// ---------------------------------------------------------------- launch
extern "C" void kernel_launch(void* const* d_in, const int* in_sizes, int n_in,
                              void* d_out, int out_size, void* d_ws, size_t ws_size,
                              hipStream_t stream) {
  (void)in_sizes; (void)n_in; (void)out_size; (void)ws_size;
  const float* hs  = (const float*)d_in[0];
  const float* fc  = (const float*)d_in[1];
  const float* fs  = (const float*)d_in[2];
  const float* wq  = (const float*)d_in[3];
  const float* bq  = (const float*)d_in[4];
  const float* wk  = (const float*)d_in[5];
  const float* bk  = (const float*)d_in[6];
  const float* wv  = (const float*)d_in[7];
  const float* bv  = (const float*)d_in[8];
  const float* wo  = (const float*)d_in[9];
  const float* bo  = (const float*)d_in[10];
  const float* nqw = (const float*)d_in[11];
  const float* nkw = (const float*)d_in[12];

  char* ws = (char*)d_ws;
  const size_t E = (size_t)S_LEN * DIMN;        // 4M elements
  u16*   Xb  = (u16*)(ws + 0);                  // bf16, 2E bytes
  u16*   Wqb = (u16*)(ws + 2  * E);
  u16*   Wkb = (u16*)(ws + 4  * E);
  u16*   Wvb = (u16*)(ws + 6  * E);
  u16*   Wob = (u16*)(ws + 8  * E);
  float* Qf  = (float*)(ws + 10 * E);           // f32, 4E bytes
  float* Kf  = (float*)(ws + 14 * E);
  float* Vf  = (float*)(ws + 18 * E);
  u16*   Qb  = (u16*)(ws + 22 * E);
  u16*   Kbb = (u16*)(ws + 24 * E);
  u16*   Vtb = (u16*)(ws + 26 * E);
  u16*   Ob  = (u16*)(ws + 28 * E);             // end at 30E = ~126 MB

  const int nE = (int)E;
  const int castGrid = nE / (8 * 256);
  k_cast_bf16<<<castGrid, 256, 0, stream>>>(hs, Xb,  nE);
  k_cast_bf16<<<castGrid, 256, 0, stream>>>(wq, Wqb, nE);
  k_cast_bf16<<<castGrid, 256, 0, stream>>>(wk, Wkb, nE);
  k_cast_bf16<<<castGrid, 256, 0, stream>>>(wv, Wvb, nE);
  k_cast_bf16<<<castGrid, 256, 0, stream>>>(wo, Wob, nE);

  dim3 gg(DIMN / 128, S_LEN / 128);
  k_gemm_bf16_nt<<<gg, 256, 0, stream>>>(Xb, Wqb, bq, Qf, S_LEN, DIMN, DIMN);
  k_gemm_bf16_nt<<<gg, 256, 0, stream>>>(Xb, Wkb, bk, Kf, S_LEN, DIMN, DIMN);
  k_gemm_bf16_nt<<<gg, 256, 0, stream>>>(Xb, Wvb, bv, Vf, S_LEN, DIMN, DIMN);

  k_rmsnorm_rope<<<S_LEN, 256, 0, stream>>>(Qf, nqw, fc, fs, Qb);
  k_rmsnorm_rope<<<S_LEN, 256, 0, stream>>>(Kf, nkw, fc, fs, Kbb);
  k_vtrans<<<dim3(2, 32, 16), 256, 0, stream>>>(Vf, Vtb);

  k_flash<<<dim3(S_LEN / 64, NH), 128, 0, stream>>>(Qb, Kbb, Vtb, Ob);

  k_gemm_bf16_nt<<<gg, 256, 0, stream>>>(Ob, Wob, bo, (float*)d_out, S_LEN, DIMN, DIMN);
}
